// RPN_66400194396317
// MI455X (gfx1250) — matrix operand order repaired and checked by reading, not compiled, not run on hardware
//
#include <hip/hip_runtime.h>
#include <math.h>

typedef __attribute__((ext_vector_type(16))) __bf16       v16bf;
typedef __attribute__((ext_vector_type(8)))  float        v8f;
typedef __attribute__((ext_vector_type(8)))  unsigned int v8u;
typedef __attribute__((ext_vector_type(4)))  int          v4i;

#define AS1 __attribute__((address_space(1)))
#define AS3 __attribute__((address_space(3)))

// Probe-confirmed on this toolchain (round 2): global_load_async_to_lds_b32/b128,
// s_wait_asynccnt, s_wait_tensorcnt, ds_load_tr16_b128_v8{i16,f16,bf16},
// global_load_tr16_b128_v8i16, tensor_load_to_lds/store_from_lds, cluster_id_x,
// s_cluster_barrier, ds_atomic_async_barrier_arrive_b64.
#if __has_builtin(__builtin_amdgcn_global_load_async_to_lds_b128)
#define HAVE_ASYNC 1
#else
#define HAVE_ASYNC 0
#endif
#if __has_builtin(__builtin_amdgcn_s_wait_asynccnt)
#define HAVE_WAIT_ASYNC 1
#else
#define HAVE_WAIT_ASYNC 0
#endif

#define NANCH      64512
#define PRE_NMS_K  2000
#define POST_NMS_K 1000
#define MCAND      6000
#define NMS_TH     0.7f
#define BBOX_CLAMP 4.135166556742356f

__device__ __forceinline__ unsigned short f2bf(float f) {
    unsigned int u = __float_as_uint(f);
    u += 0x7FFFu + ((u >> 16) & 1u);        // round-to-nearest-even
    return (unsigned short)(u >> 16);
}
__device__ __forceinline__ float bf2f(unsigned int hs) {
    return __uint_as_float(hs << 16);
}
__device__ __forceinline__ unsigned okey(float f) {  // order-preserving key
    unsigned u = __float_as_uint(f);
    return (u & 0x80000000u) ? ~u : (u | 0x80000000u);
}

// ---------------------------------------------------------------- converts
__global__ void k_f32_to_bf16(const float* __restrict__ src,
                              unsigned short* __restrict__ dst, int n) {
    int i = blockIdx.x * blockDim.x + threadIdx.x;
    if (i < n) dst[i] = f2bf(src[i]);
}

// conv_w [co=256][ci=256][3][3] -> wp[tap r][ci][co] bf16
__global__ void k_pack_w(const float* __restrict__ w,
                         unsigned short* __restrict__ wp) {
    int t = blockIdx.x * blockDim.x + threadIdx.x;
    if (t >= 9 * 256 * 256) return;
    int co = t & 255, ci = (t >> 8) & 255, r = t >> 16;
    wp[t] = f2bf(w[(co * 256 + ci) * 9 + r]);
}

// ---------------------------------------------------------------- 3x3 conv, implicit GEMM via WMMA bf16
// Block: 256 thr = 8 waves; tile = 64 pixels x 32 co.
// Per dy (3): stage an aligned 80-pixel x 32-ci window once (async b128 -> LDS);
// dx handled by an LDS offset at gather time. Boundary zeroing folded into a
// per-lane validity mask on the A fragment (one lane owns one output pixel).
#define AHS  88   // ldsA row stride (ushorts): 80 window + 8 pad (keeps 16B chunks aligned)
#define APAD 8    // left halo inside the window
#define BHS  34
__global__ __launch_bounds__(256) void k_conv3x3_wmma(
    const unsigned short* __restrict__ feat,   // [B][256][H][W] bf16
    const unsigned short* __restrict__ wp,     // [9][256][256]  bf16
    const float* __restrict__ bias,            // [256]
    unsigned short* __restrict__ hout,         // [B][H*W][256]  bf16
    int H, int W) {
    __shared__ unsigned short ldsA[32 * AHS];       // [ci][80-px window]
    __shared__ unsigned short ldsB[3 * 32 * BHS];   // [dx][co][ci]
    const int HW = H * W;
    const int tid = threadIdx.x;
    const int b = blockIdx.y;
    const int tileBase = blockIdx.x * 64;
    const int cob = blockIdx.z * 32;
    const int lane = tid & 31, wid = tid >> 5;
    const int pg = wid & 3, cg = wid >> 2;     // wave -> (pixel group, co group)
    const int mn = lane & 15, half = lane >> 4;
    const int pl = pg * 16 + mn;               // this lane's pixel within the tile
    const int p0 = tileBase + pl;
    const int px0 = p0 % W, py0 = p0 / W;

    v8f acc = {};

    for (int dy = -1; dy <= 1; ++dy) {
        const int rowShift = dy * W;
        for (int cib = 0; cib < 8; ++cib) {
            __syncthreads();
            // ---- stage A window: 32 ci x 80 px, one aligned b128 chunk per unit
            for (int c = tid; c < 320; c += 256) {
                int ci = c / 10, ch = c % 10;
                int g0 = tileBase + rowShift - APAD + ch * 8;
                const size_t plane = (size_t)(b * 256 + cib * 32 + ci) * HW;
                unsigned short* dst = &ldsA[ci * AHS + ch * 8];
                if (g0 >= 0 && g0 + 8 <= HW) {
                    const unsigned short* src = feat + plane + g0;
#if HAVE_ASYNC
                    __builtin_amdgcn_global_load_async_to_lds_b128(
                        (AS1 v4i*)src, (AS3 v4i*)dst, 0, 0);
#else
                    *(uint4*)dst = *(const uint4*)src;
#endif
                } else {
                    // image-edge chunk: per-element clamp (values masked at compute)
#pragma unroll
                    for (int q = 0; q < 8; ++q) {
                        int gq = g0 + q;
                        gq = gq < 0 ? 0 : (gq >= HW ? HW - 1 : gq);
                        dst[q] = feat[plane + gq];
                    }
                }
            }
            // ---- stage B: 3 taps x 32 co x 32 ci (transposed to [co][ci])
            for (int e = tid; e < 3072; e += 256) {
                int dxi = e >> 10, rem = e & 1023;
                int co = rem & 31, ci = rem >> 5;
                ldsB[(dxi * 32 + co) * BHS + ci] =
                    wp[((size_t)((dy + 1) * 3 + dxi) * 256 + cib * 32 + ci) * 256 +
                       cob + co];
            }
#if HAVE_ASYNC
#if HAVE_WAIT_ASYNC
            __builtin_amdgcn_s_wait_asynccnt(0);
#else
            asm volatile("s_wait_asynccnt 0" ::: "memory");
#endif
#endif
            __syncthreads();

            // ---- 3 taps of WMMA from the shared window
#pragma unroll
            for (int dxi = 0; dxi < 3; ++dxi) {
                const int dx = dxi - 1;
                const bool lv = (px0 + dx >= 0) && (px0 + dx < W) &&
                                (py0 + dy >= 0) && (py0 + dy < H);
                v8u au, bu;
                const unsigned short* ra = &ldsA[APAD + pl + dx];
                const unsigned short* rb = &ldsB[(dxi * 32 + cg * 16 + mn) * BHS];
#pragma unroll
                for (int jj = 0; jj < 8; ++jj) {
                    int ka = 2 * jj + ((jj >= 4) ? 8 : 0) + 8 * half;  // A 16x32 packing
                    unsigned lo = ra[ka * AHS];
                    unsigned hi = ra[(ka + 1) * AHS];
                    au[jj] = lv ? (lo | (hi << 16)) : 0u;
                    int kb = 2 * jj + ((jj >= 4) ? 8 : 0) + 8 * half;  // B 32x16 packing
                    bu[jj] = *(const unsigned int*)(rb + kb);
                }
                acc = __builtin_amdgcn_wmma_f32_16x16x32_bf16(
                    false, __builtin_bit_cast(v16bf, au),
                    false, __builtin_bit_cast(v16bf, bu),
                    (short)0, acc, false, false);
            }
        }
    }
    // epilogue: bias + relu, store bf16, [B][pix][256]
#pragma unroll
    for (int rr = 0; rr < 8; ++rr) {
        int m = rr + 8 * half;
        int px = tileBase + pg * 16 + m;
        int co = cob + cg * 16 + mn;
        float v = acc[rr] + bias[co];
        v = v > 0.f ? v : 0.f;
        hout[((size_t)b * HW + px) * 256 + co] = f2bf(v);
    }
}

// ---------------------------------------------------------------- 1x1 heads + sigmoid + decode + clip
__global__ __launch_bounds__(256) void k_head_decode(
    const unsigned short* __restrict__ hbuf,   // [B][HW][256] bf16
    const float* __restrict__ cls_w, const float* __restrict__ cls_b,
    const float* __restrict__ reg_w, const float* __restrict__ reg_b,
    float* __restrict__ scores,                // [B][NANCH]
    float4* __restrict__ boxes,                // [B][NANCH]
    int H, int W, int stride, float asize, int aoff) {
    __shared__ float wls[15 * 256];
    __shared__ float bls[15];
    int tid = threadIdx.x;
    for (int i = tid; i < 15 * 256; i += 256) {
        int o = i >> 8, ci = i & 255;
        wls[i] = (o < 3) ? cls_w[o * 256 + ci] : reg_w[(o - 3) * 256 + ci];
    }
    if (tid < 15) bls[tid] = (tid < 3) ? cls_b[tid] : reg_b[tid - 3];
    __syncthreads();

    int HW = H * W;
    int idx = blockIdx.x * 256 + tid;
    if (idx >= 4 * HW) return;
    int b = idx / HW, pp = idx % HW;
    const unsigned int* hp =
        (const unsigned int*)(hbuf + ((size_t)b * HW + pp) * 256);

    float acc[15];
#pragma unroll
    for (int o = 0; o < 15; ++o) acc[o] = bls[o];
    for (int c2 = 0; c2 < 128; ++c2) {
        unsigned int u = hp[c2];
        float h0 = bf2f(u & 0xFFFFu), h1 = bf2f(u >> 16);
        int ci = c2 * 2;
#pragma unroll
        for (int o = 0; o < 15; ++o)
            acc[o] += h0 * wls[o * 256 + ci] + h1 * wls[o * 256 + ci + 1];
    }

    int y = pp / W, x = pp % W;
    float sx = (float)(x * stride), sy = (float)(y * stride);
#pragma unroll
    for (int a = 0; a < 3; ++a) {
        float ratio = (a == 0) ? 0.5f : ((a == 1) ? 1.0f : 2.0f);
        float hh = sqrtf(ratio), ww = 1.0f / hh;
        float ax1 = sx + rintf(-asize * ww * 0.5f);
        float ay1 = sy + rintf(-asize * hh * 0.5f);
        float ax2 = sx + rintf(asize * ww * 0.5f);
        float ay2 = sy + rintf(asize * hh * 0.5f);
        float wa = ax2 - ax1, ha = ay2 - ay1;
        float cxa = ax1 + 0.5f * wa, cya = ay1 + 0.5f * ha;
        float ddx = acc[3 + a * 4 + 0], ddy = acc[3 + a * 4 + 1];
        float ddw = fminf(acc[3 + a * 4 + 2], BBOX_CLAMP);
        float ddh = fminf(acc[3 + a * 4 + 3], BBOX_CLAMP);
        float cx = ddx * wa + cxa, cy = ddy * ha + cya;
        float bw = expf(ddw) * wa, bh = expf(ddh) * ha;
        float x1 = fminf(fmaxf(cx - 0.5f * bw, 0.f), 1024.f);
        float y1 = fminf(fmaxf(cy - 0.5f * bh, 0.f), 1024.f);
        float x2 = fminf(fmaxf(cx + 0.5f * bw, 0.f), 1024.f);
        float y2 = fminf(fmaxf(cy + 0.5f * bh, 0.f), 1024.f);
        bool valid = (x2 - x1 >= 1e-3f) && (y2 - y1 >= 1e-3f);
        float s = 1.0f / (1.0f + expf(-acc[a]));
        int gi = aoff + pp * 3 + a;
        scores[(size_t)b * NANCH + gi] = valid ? s : -__builtin_inff();
        boxes[(size_t)b * NANCH + gi] = make_float4(x1, y1, x2, y2);
    }
}

// ---------------------------------------------------------------- per (batch,level) top-2000 via radix select
__global__ __launch_bounds__(1024) void k_topk(
    const float* __restrict__ scores,
    float* __restrict__ candS, int* __restrict__ candI) {
    const int LN[3] = {49152, 12288, 3072};
    const int LB[3] = {0, 49152, 61440};
    int blk = blockIdx.x;                 // b*3 + lvl
    int b = blk / 3, lvl = blk % 3;
    int n = LN[lvl], base = LB[lvl];
    const float* sc = scores + (size_t)b * NANCH + base;
    int tid = threadIdx.x;
    __shared__ unsigned int scnt;

    unsigned prefix = 0;
    for (int bit = 31; bit >= 0; --bit) {
        unsigned test = prefix | (1u << bit);
        if (tid == 0) scnt = 0;
        __syncthreads();
        unsigned local = 0;
        for (int i = tid; i < n; i += 1024)
            if (okey(sc[i]) >= test) local++;
        atomicAdd(&scnt, local);
        __syncthreads();
        if (scnt >= PRE_NMS_K) prefix = test;
        __syncthreads();
    }
    if (tid == 0) scnt = 0;
    __syncthreads();
    float* cs = candS + (size_t)blk * PRE_NMS_K;
    int* ci = candI + (size_t)blk * PRE_NMS_K;
    for (int i = tid; i < n; i += 1024) {
        if (okey(sc[i]) > prefix) {
            unsigned p = atomicAdd(&scnt, 1u);
            if (p < PRE_NMS_K) { cs[p] = sc[i]; ci[p] = base + i; }
        }
    }
    __syncthreads();
    for (int i = tid; i < n; i += 1024) {
        if (okey(sc[i]) == prefix) {
            unsigned p = atomicAdd(&scnt, 1u);
            if (p < PRE_NMS_K) { cs[p] = sc[i]; ci[p] = base + i; }
        }
    }
}

// ---------------------------------------------------------------- per-batch sort + greedy NMS + emit
__global__ __launch_bounds__(1024) void k_nms(
    const float* __restrict__ candS, const int* __restrict__ candI,
    const float4* __restrict__ boxes,
    float4* __restrict__ sb, float4* __restrict__ nb,
    float* __restrict__ area, float* __restrict__ out) {
    extern __shared__ unsigned char smemRaw[];
    float* sKey = (float*)smemRaw;          // 8192 (= -score; ascending sort)
    int* sIdx = (int*)(sKey + 8192);        // 8192
    __shared__ unsigned char alive[MCAND];
    __shared__ int tsum[1024];
    int b = blockIdx.x, tid = threadIdx.x;

    for (int i = tid; i < 8192; i += 1024) {
        if (i < MCAND) {
            sKey[i] = -candS[(size_t)b * MCAND + i];
            sIdx[i] = candI[(size_t)b * MCAND + i];
        } else { sKey[i] = __builtin_inff(); sIdx[i] = 0; }
    }
    __syncthreads();

    for (int k = 2; k <= 8192; k <<= 1) {
        for (int j = k >> 1; j > 0; j >>= 1) {
            for (int t = tid; t < 4096; t += 1024) {
                int i = 2 * j * (t / j) + (t % j);
                int ixj = i + j;
                bool up = ((i & k) == 0);
                float a = sKey[i], c = sKey[ixj];
                if ((a > c) == up) {
                    sKey[i] = c; sKey[ixj] = a;
                    int ti = sIdx[i]; sIdx[i] = sIdx[ixj]; sIdx[ixj] = ti;
                }
            }
            __syncthreads();
        }
    }

    size_t b6 = (size_t)b * MCAND;
    for (int i = tid; i < MCAND; i += 1024) {
        int gi = sIdx[i];
        float4 bx = boxes[(size_t)b * NANCH + gi];
        int lvl = (gi < 49152) ? 0 : ((gi < 61440) ? 1 : 2);
        float off = lvl * 1025.0f;
        float4 nbx = make_float4(bx.x + off, bx.y + off, bx.z + off, bx.w + off);
        sb[b6 + i] = bx;
        nb[b6 + i] = nbx;
        area[b6 + i] = (nbx.z - nbx.x) * (nbx.w - nbx.y);
        float sk = sKey[i];
        alive[i] = (sk > -3e38f && sk < 3e38f) ? 1 : 0;
    }
    __syncthreads();

    for (int i = 0; i < MCAND; ++i) {
        __syncthreads();
        if (!alive[i]) continue;
        float4 bi = nb[b6 + i];
        float ai = area[b6 + i];
        for (int j = i + 1 + tid; j < MCAND; j += 1024) {
            if (!alive[j]) continue;
            float4 bj = nb[b6 + j];
            float lx = fmaxf(bi.x, bj.x), ly = fmaxf(bi.y, bj.y);
            float rx = fminf(bi.z, bj.z), ry = fminf(bi.w, bj.w);
            float iw = fmaxf(rx - lx, 0.f), ih = fmaxf(ry - ly, 0.f);
            float inter = iw * ih;
            float iou = inter / (ai + area[b6 + j] - inter);
            if (iou > NMS_TH) alive[j] = 0;
        }
    }
    __syncthreads();

    const int CH = 6;
    int beg = tid * CH;
    int cnt = 0;
    for (int ii = beg; ii < beg + CH && ii < MCAND; ++ii) cnt += alive[ii];
    tsum[tid] = cnt;
    __syncthreads();
    if (tid == 0) {
        int run = 0;
        for (int t = 0; t < 1024; ++t) { int c = tsum[t]; tsum[t] = run; run += c; }
    }
    __syncthreads();
    float* ob = out + (size_t)b * POST_NMS_K * 5;
    for (int t = tid; t < POST_NMS_K * 5; t += 1024) ob[t] = 0.f;
    __syncthreads();
    int r = tsum[tid];
    for (int ii = beg; ii < beg + CH && ii < MCAND; ++ii) {
        if (alive[ii]) {
            if (r < POST_NMS_K) {
                float4 bx = sb[b6 + ii];
                ob[r * 5 + 0] = bx.x; ob[r * 5 + 1] = bx.y;
                ob[r * 5 + 2] = bx.z; ob[r * 5 + 3] = bx.w;
                ob[r * 5 + 4] = -sKey[ii];
            }
            r++;
        }
    }
}

// ---------------------------------------------------------------- host
extern "C" void kernel_launch(void* const* d_in, const int* in_sizes, int n_in,
                              void* d_out, int out_size, void* d_ws, size_t ws_size,
                              hipStream_t stream) {
    (void)in_sizes; (void)n_in; (void)out_size; (void)ws_size;
    const float* feat[3] = {(const float*)d_in[1], (const float*)d_in[2],
                            (const float*)d_in[3]};
    const float* conv_w = (const float*)d_in[4];
    const float* conv_b = (const float*)d_in[5];
    const float* cls_w  = (const float*)d_in[6];
    const float* cls_b  = (const float*)d_in[7];
    const float* reg_w  = (const float*)d_in[8];
    const float* reg_b  = (const float*)d_in[9];
    float* out = (float*)d_out;

    const int Hs[3] = {128, 64, 32};
    const int strides[3] = {8, 16, 32};
    const float sizes[3] = {64.f, 128.f, 256.f};
    const int aoff[3] = {0, 49152, 61440};

    char* ws = (char*)d_ws;
    size_t off = 0;
    auto take = [&](size_t bytes) -> char* {
        char* p = ws + off;
        off = (off + bytes + 255) & ~(size_t)255;
        return p;
    };
    unsigned short* featBF[3];
    unsigned short* hbuf[3];
    for (int l = 0; l < 3; ++l)
        featBF[l] = (unsigned short*)take((size_t)4 * 256 * Hs[l] * Hs[l] * 2);
    unsigned short* wp = (unsigned short*)take((size_t)9 * 256 * 256 * 2);
    for (int l = 0; l < 3; ++l)
        hbuf[l] = (unsigned short*)take((size_t)4 * Hs[l] * Hs[l] * 256 * 2);
    float*  scores = (float*)take((size_t)4 * NANCH * 4);
    float4* boxes  = (float4*)take((size_t)4 * NANCH * 16);
    float*  candS  = (float*)take((size_t)4 * MCAND * 4);
    int*    candI  = (int*)take((size_t)4 * MCAND * 4);
    float4* sb     = (float4*)take((size_t)4 * MCAND * 16);
    float4* nbuf   = (float4*)take((size_t)4 * MCAND * 16);
    float*  area   = (float*)take((size_t)4 * MCAND * 4);

    for (int l = 0; l < 3; ++l) {
        int n = 4 * 256 * Hs[l] * Hs[l];
        k_f32_to_bf16<<<(n + 255) / 256, 256, 0, stream>>>(feat[l], featBF[l], n);
    }
    {
        int n = 9 * 256 * 256;
        k_pack_w<<<(n + 255) / 256, 256, 0, stream>>>(conv_w, wp);
    }
    for (int l = 0; l < 3; ++l) {
        int H = Hs[l], W = Hs[l], HW = H * W;
        dim3 g(HW / 64, 4, 8);
        k_conv3x3_wmma<<<g, 256, 0, stream>>>(featBF[l], wp, conv_b, hbuf[l], H, W);
    }
    for (int l = 0; l < 3; ++l) {
        int H = Hs[l], W = Hs[l], HW = H * W;
        int tot = 4 * HW;
        k_head_decode<<<(tot + 255) / 256, 256, 0, stream>>>(
            hbuf[l], cls_w, cls_b, reg_w, reg_b, scores, boxes,
            H, W, strides[l], sizes[l], aoff[l]);
    }
    k_topk<<<12, 1024, 0, stream>>>(scores, candS, candI);
    k_nms<<<4, 1024, 65536, stream>>>(candS, candI, boxes, sb, nbuf, area, out);
}
